// DSQGAttentionV5_86139864089302
// MI455X (gfx1250) — compile-verified, matrix operand order checked by reading, not run on hardware
//
#include <hip/hip_runtime.h>
#include <math.h>

typedef __attribute__((ext_vector_type(16))) _Float16 v16h;
typedef __attribute__((ext_vector_type(8)))  float    v8f;

#define WAVES   8
#define TQ      16
#define BQ      (WAVES * TQ)   // 128 queries per block
#define HDIM    64
#define NOFF    44
#define NDENSE  33
#define NSPARSE 11

__device__ __forceinline__ v8f wmma_f16(v16h a, v16h b, v8f c) {
    // D = A(16x32 f16) * B(32x16 f16) + C(16x16 f32)
    return __builtin_amdgcn_wmma_f32_16x16x32_f16(
        /*neg_a=*/false, a, /*neg_b=*/false, b,
        /*c_mod=*/(short)0, c, /*reuse_a=*/false, /*reuse_b=*/false);
}

// Pack 16 f32 values of one contiguous row into the CDNA5 16-bit A/B lane
// layout for one K=32 chunk: elements e=0..7 -> src[base+e],
// e=8..15 -> src[base+16+e-8]  (base = chunk*32 + (lane>=16)*8).
__device__ __forceinline__ v16h pack_contig(const float* __restrict__ src, int base) {
    v16h r;
#pragma unroll
    for (int e = 0; e < 8; ++e) r[e]     = (_Float16)src[base + e];
#pragma unroll
    for (int e = 0; e < 8; ++e) r[e + 8] = (_Float16)src[base + 16 + e];
    return r;
}

__global__ __launch_bounds__(256)
void dsqg_attn_v5_kernel(const float* __restrict__ q,  const float* __restrict__ k,
                         const float* __restrict__ v,  const float* __restrict__ pb,
                         const float* __restrict__ se, const float* __restrict__ phase_base,
                         const float* __restrict__ phase_gain,
                         const float* __restrict__ y_pre, const float* __restrict__ z_pre,
                         float* __restrict__ out, int Nlen)
{
    constexpr int OFFS[NOFF] = {
        0,1,2,3,4,5,6,7,8,9,10,11,12,13,14,15,16,17,18,19,20,21,22,23,24,25,
        26,27,28,29,30,31,32, 48,64,96,128,192,256,384,512,768,1024,1536};
    constexpr float SC = 0.125f;  // 1/sqrt(64)

    __shared__ float s_qse[WAVES][TQ][48];  // q . se_i   per query per offset
    __shared__ float s_sc [WAVES][TQ][48];  // raw q.k scores, then alpha (in place)

    const int tid  = threadIdx.x;
    const int w    = tid >> 5;
    const int lane = tid & 31;
    const int nrow = lane & 15;   // A-row / B-col / C-col index
    const int hl   = lane >> 4;   // which K-half (A/B) / row-half (C)

    const int h  = blockIdx.y;
    const int b  = blockIdx.z;
    const int bh = b * gridDim.y + h;
    const int n0 = blockIdx.x * BQ + w * TQ;   // first query of this wave's tile

    const float* qb = q + (size_t)bh * Nlen * HDIM;
    const float* kb = k + (size_t)bh * Nlen * HDIM;
    const float* vb = v + (size_t)bh * Nlen * HDIM;

    // hint the dense K window into cache early
    {
        int p = n0 - 32; if (p < 0) p = 0;
        __builtin_prefetch((const void*)(kb + (size_t)p * HDIM), 0, 3);
    }

    // ---- pack this wave's Q tile into WMMA A layout (two K=32 chunks) ----
    const int pbase = hl * 8;
    const float* qrow = qb + (size_t)(n0 + nrow) * HDIM;
    const v16h aq0 = pack_contig(qrow, pbase);
    const v16h aq1 = pack_contig(qrow, 32 + pbase);

    // ---- q . se_i for all 44 offsets via WMMA (se rows act as keys) ----
#pragma unroll
    for (int t = 0; t < 3; ++t) {
        int i = t * 16 + nrow; if (i > NOFF - 1) i = NOFF - 1;
        const float* srow = se + (size_t)i * HDIM;
        v16h b0 = pack_contig(srow, pbase);
        v16h b1 = pack_contig(srow, 32 + pbase);
        v8f c = {};
        c = wmma_f16(aq0, b0, c);
        c = wmma_f16(aq1, b1, c);
#pragma unroll
        for (int r = 0; r < 8; ++r) {
            int m  = r + hl * 8;
            int ii = t * 16 + nrow;
            if (ii < NOFF) s_qse[w][m][ii] = c[r];
        }
    }

    // ---- q . k scores: 3 dense tiles (48-key band) + 11 sparse diagonal tiles ----
#pragma unroll
    for (int tt = 0; tt < 3 + NSPARSE; ++tt) {
        const int kbrel = (tt < 3) ? (-32 + 16 * tt) : -OFFS[NDENSE + (tt - 3)];
        int key = n0 + kbrel + nrow; if (key < 0) key = 0;  // masked later
        const float* krow = kb + (size_t)key * HDIM;
        v16h b0 = pack_contig(krow, pbase);
        v16h b1 = pack_contig(krow, 32 + pbase);
        v8f c = {};
        c = wmma_f16(aq0, b0, c);
        c = wmma_f16(aq1, b1, c);
        if (tt < 3) {
#pragma unroll
            for (int r = 0; r < 8; ++r) {
                int m = r + hl * 8;
                int d = m + 32 - 16 * tt - nrow;     // offset index for (m, key col)
                if (d >= 0 && d < NDENSE) s_sc[w][m][d] = c[r];
            }
        } else {
            // only the diagonal (col == row) carries score for offset d
            if ((nrow >> 3) == hl) {
#pragma unroll
                for (int r = 0; r < 8; ++r)
                    if (r == (nrow & 7)) s_sc[w][nrow][NDENSE + (tt - 3)] = c[r];
            }
        }
    }
    __syncthreads();

    // ---- per-query 44-way masked softmax (lanes 0..15 own one query each) ----
    if (lane < 16) {
        const int m  = nrow;
        const int ng = n0 + m;
        float sv[NOFF];
        float mx = -INFINITY;
#pragma unroll
        for (int i = 0; i < NOFF; ++i) {
            float s = (ng >= OFFS[i])
                    ? (s_sc[w][m][i] + s_qse[w][m][i]) * SC + pb[i * 16 + h]
                    : -INFINITY;
            sv[i] = s;
            mx = fmaxf(mx, s);
        }
        float sum = 0.f;
#pragma unroll
        for (int i = 0; i < NOFF; ++i) { float e = __expf(sv[i] - mx); sv[i] = e; sum += e; }
        const float rs = 1.f / sum;
#pragma unroll
        for (int i = 0; i < NOFF; ++i) s_sc[w][m][i] = sv[i] * rs;   // alpha in place
    }
    __syncthreads();

    // ---- dense PV: banded 16x48 alpha (f16) x V window (48x64) via WMMA ----
    v16h aa0, aa1;
#pragma unroll
    for (int e = 0; e < 16; ++e) {
        const int j0 = (e & 7) + ((e >> 3) << 4) + (hl << 3);   // key col in chunk0
        const int d0 = nrow + 32 - j0;
        aa0[e] = (d0 >= 0 && d0 < NDENSE) ? (_Float16)s_sc[w][nrow][d0] : (_Float16)0.f;
        const int j1 = 32 + j0;                                  // key col in chunk1
        const int d1 = nrow + 32 - j1;
        aa1[e] = (j1 < 48 && d1 >= 0) ? (_Float16)s_sc[w][nrow][d1] : (_Float16)0.f;
    }

    v8f acc[4];
#pragma unroll
    for (int tn = 0; tn < 4; ++tn) {
        const int col = tn * 16 + nrow;
        v16h b0, b1;
#pragma unroll
        for (int e = 0; e < 16; ++e) {
            const int j = (e & 7) + ((e >> 3) << 4) + (hl << 3);
            int k0 = n0 - 32 + j;      if (k0 < 0) k0 = 0;
            b0[e] = (_Float16)vb[(size_t)k0 * HDIM + col];
            const int j1 = 32 + j;
            int k1 = n0 - 32 + j1;     if (k1 < 0) k1 = 0;
            b1[e] = (j1 < 48) ? (_Float16)vb[(size_t)k1 * HDIM + col] : (_Float16)0.f;
        }
        v8f c = {};
        c = wmma_f16(aa0, b0, c);
        c = wmma_f16(aa1, b1, c);
        acc[tn] = c;
    }

    // ---- sparse PV with per-(query,key) phase rotation of v[0..3] (VALU) ----
#pragma unroll
    for (int si = 0; si < NSPARSE; ++si) {
        const int d = OFFS[NDENSE + si];
        const float pb0 = phase_base[((size_t)si * 16 + h) * 2 + 0];
        const float pb1 = phase_base[((size_t)si * 16 + h) * 2 + 1];
        const float pg0 = phase_gain[((size_t)si * 16 + h) * 2 + 0];
        const float pg1 = phase_gain[((size_t)si * 16 + h) * 2 + 1];
#pragma unroll
        for (int r = 0; r < 8; ++r) {
            const int m  = r + hl * 8;
            const int ng = n0 + m;
            if (ng < d) continue;                     // masked: alpha == 0
            const float alpha = s_sc[w][m][NDENSE + si];
            const int key = ng - d;
            const float y0 = y_pre[((size_t)bh * Nlen + ng) * 2 + 0];
            const float y1 = y_pre[((size_t)bh * Nlen + ng) * 2 + 1];
            const float z0 = z_pre[((size_t)bh * Nlen + key) * 2 + 0];
            const float z1 = z_pre[((size_t)bh * Nlen + key) * 2 + 1];
            float c0, s0, c1, s1;
            __sincosf(pb0 + pg0 * y0 * z0, &s0, &c0);
            __sincosf(pb1 + pg1 * y1 * z1, &s1, &c1);
            const float* vr = vb + (size_t)key * HDIM;
#pragma unroll
            for (int tn = 0; tn < 4; ++tn) {
                const int col = tn * 16 + nrow;
                const float vv = vr[col];
                float add;
                if      (col == 0) add = c0 * vv - s0 * vr[1];
                else if (col == 1) add = s0 * vr[0] + c0 * vv;
                else if (col == 2) add = c1 * vv - s1 * vr[3];
                else if (col == 3) add = s1 * vr[2] + c1 * vv;
                else               add = vv;
                acc[tn][r] += alpha * add;
            }
        }
    }

    // ---- store (C layout: row = r + 8*hl, col = tn*16 + nrow) ----
    float* ob = out + (size_t)bh * Nlen * HDIM;
#pragma unroll
    for (int tn = 0; tn < 4; ++tn)
#pragma unroll
        for (int r = 0; r < 8; ++r)
            ob[(size_t)(n0 + r + hl * 8) * HDIM + tn * 16 + nrow] = acc[tn][r];
}

extern "C" void kernel_launch(void* const* d_in, const int* in_sizes, int n_in,
                              void* d_out, int out_size, void* d_ws, size_t ws_size,
                              hipStream_t stream) {
    const float* q          = (const float*)d_in[0];
    const float* k          = (const float*)d_in[1];
    const float* v          = (const float*)d_in[2];
    const float* pb         = (const float*)d_in[3];
    const float* se         = (const float*)d_in[4];
    const float* phase_base = (const float*)d_in[5];
    const float* phase_gain = (const float*)d_in[6];
    const float* y_pre      = (const float*)d_in[7];
    const float* z_pre      = (const float*)d_in[8];
    float* out              = (float*)d_out;

    const int B = 2, H = 16;
    const int Nlen = in_sizes[0] / (B * H * HDIM);   // 4096 for the reference shapes

    dim3 grid(Nlen / BQ, H, B);
    dim3 block(WAVES * 32, 1, 1);
    dsqg_attn_v5_kernel<<<grid, block, 0, stream>>>(
        q, k, v, pb, se, phase_base, phase_gain, y_pre, z_pre, out, Nlen);
    (void)n_in; (void)out_size; (void)d_ws; (void)ws_size;
}
